// AFiReHead_68513318305868
// MI455X (gfx1250) — matrix operand order repaired and checked by reading, not compiled, not run on hardware
//
#include <hip/hip_runtime.h>
#include <hip/hip_bf16.h>
#include <math.h>

typedef __bf16 bf16;
typedef __attribute__((ext_vector_type(16))) __bf16 bf16x16;
typedef __attribute__((ext_vector_type(8)))  __bf16 bf16x8;
typedef __attribute__((ext_vector_type(4)))  __bf16 bf16x4;
typedef __attribute__((ext_vector_type(8)))  float  f32x8;

#define WMMA_BF16(a, b, c) \
  __builtin_amdgcn_wmma_f32_16x16x32_bf16(false, (a), false, (b), (short)0, (c), false, false)

// Problem dimensions (fixed by the reference)
#define BNUM  64
#define PNUM  196
#define MROWS (BNUM * PNUM)   // 12544
#define DIN   768
#define DH    1024
#define DBT   256
#define DOUT  4096

// ---------------------------------------------------------------------------
// fp32 -> bf16 elementwise convert (vector of 4 per thread)
// ---------------------------------------------------------------------------
__global__ __launch_bounds__(256) void cvt_f32_bf16(const float* __restrict__ in,
                                                    bf16* __restrict__ out, int n4) {
  int i = blockIdx.x * blockDim.x + threadIdx.x;
  if (i < n4) {
    float4 f = ((const float4*)in)[i];
    bf16x4 o;
    o[0] = (bf16)f.x; o[1] = (bf16)f.y; o[2] = (bf16)f.z; o[3] = (bf16)f.w;
    ((bf16x4*)out)[i] = o;
  }
}

// ---------------------------------------------------------------------------
// Tiled WMMA GEMM: C[M,N] = act(A[M,K](bf16) x W[N,K](bf16)^T + bias)
// Block: 256 threads = 8 waves. Block tile 64 x 128; wave tile 64 x 16.
// ---------------------------------------------------------------------------
template <bool GELU_ACT, bool OUT_BF16>
__global__ __launch_bounds__(256) void gemm_mlp(const bf16* __restrict__ A,
                                                const bf16* __restrict__ W,
                                                const float* __restrict__ bias,
                                                void* __restrict__ Cout,
                                                int K, int N) {
  const int lane = threadIdx.x & 31;
  const int wave = threadIdx.x >> 5;
  const int row_base = blockIdx.x * 64;
  const int n = blockIdx.y * 128 + wave * 16 + (lane & 15);
  const int abase_k = (lane >> 4) << 4;  // A frag: lanes 0-15 -> K+0, 16-31 -> K+16
  const int bbase_k = (lane >> 4) << 4;  // B frag: lanes 0-15 -> K+0, 16-31 -> K+16

  f32x8 acc[4] = {};
  const bf16* wrow = W + (size_t)n * K + bbase_k;
  const bf16* arow = A + (size_t)(row_base + (lane & 15)) * K + abase_k;

  for (int k0 = 0; k0 < K; k0 += 32) {
    bf16x16 bfrag = *(const bf16x16*)(wrow + k0);  // 16 contiguous K per lane
#pragma unroll
    for (int t = 0; t < 4; ++t) {
      const bf16* at = arow + (size_t)(16 * t) * K + k0;
      bf16x8 lo = *(const bf16x8*)(at);        // K kb .. kb+7
      bf16x8 hi = *(const bf16x8*)(at + 8);    // K kb+8 .. kb+15
      bf16x16 afrag;
#pragma unroll
      for (int i = 0; i < 8; ++i) { afrag[i] = lo[i]; afrag[8 + i] = hi[i]; }
      acc[t] = WMMA_BF16(afrag, bfrag, acc[t]);
    }
  }

  const float bn = bias[n];
  const int mb = row_base + ((lane >> 4) << 3);
#pragma unroll
  for (int t = 0; t < 4; ++t) {
#pragma unroll
    for (int i = 0; i < 8; ++i) {
      float vv = acc[t][i] + bn;
      if (GELU_ACT) vv = 0.5f * vv * (1.0f + erff(vv * 0.70710678118654752f));
      const size_t idx = (size_t)(mb + 16 * t + i) * N + n;
      if (OUT_BF16) ((bf16*)Cout)[idx] = (bf16)vv;
      else          ((float*)Cout)[idx] = vv;
    }
  }
}

// ---------------------------------------------------------------------------
// Row-wise L2 normalize (rows of 256 fp32) -> bf16. One wave per row.
// ---------------------------------------------------------------------------
__global__ __launch_bounds__(256) void l2norm_rows(const float* __restrict__ Z,
                                                   bf16* __restrict__ Zn) {
  const int lane = threadIdx.x & 31;
  const int wave = threadIdx.x >> 5;
  const int row = blockIdx.x * 8 + wave;
  const float* zr = Z + (size_t)row * 256;
  float4 a = ((const float4*)zr)[lane];
  float4 b = ((const float4*)zr)[lane + 32];
  float ss = a.x * a.x + a.y * a.y + a.z * a.z + a.w * a.w +
             b.x * b.x + b.y * b.y + b.z * b.z + b.w * b.w;
#pragma unroll
  for (int off = 16; off > 0; off >>= 1) ss += __shfl_xor(ss, off, 32);
  const float s = 1.0f / fmaxf(sqrtf(ss), 1e-12f);
  bf16x4 o0, o1;
  o0[0] = (bf16)(a.x * s); o0[1] = (bf16)(a.y * s); o0[2] = (bf16)(a.z * s); o0[3] = (bf16)(a.w * s);
  o1[0] = (bf16)(b.x * s); o1[1] = (bf16)(b.y * s); o1[2] = (bf16)(b.z * s); o1[3] = (bf16)(b.w * s);
  bf16x4* dst = (bf16x4*)(Zn + (size_t)row * 256);
  dst[lane] = o0;
  dst[lane + 32] = o1;
}

// ---------------------------------------------------------------------------
// Fused weight-norm head: out[b,p,o] = g[p,o]/||v[p,o,:]|| * sum_d z[b,p,d] v[p,o,d]
// Block = (patch p, 128-wide o tile). z_p staged in padded LDS; v streamed once
// (fp32 -> bf16 in-register), sum of squares fused into the same pass.
// ---------------------------------------------------------------------------
__global__ __launch_bounds__(256) void head_gemm(const bf16* __restrict__ Zn,
                                                 const float* __restrict__ V,
                                                 const float* __restrict__ G,
                                                 float* __restrict__ Out) {
  const int p = blockIdx.x;
  const int lane = threadIdx.x & 31;
  const int wave = threadIdx.x >> 5;

  __shared__ bf16 zsh[64 * 264];  // 64 rows, stride 264 (8-elem pad -> no bank conflicts)

  {  // stage z_p (rows are strided by PNUM in Zn): thread t -> row t>>2, 64-elem segment t&3
    const int r = threadIdx.x >> 2, s = threadIdx.x & 3;
    const bf16* src = Zn + ((size_t)r * PNUM + p) * 256 + s * 64;
    bf16* dst = zsh + r * 264 + s * 64;
#pragma unroll
    for (int c = 0; c < 8; ++c) ((int4*)dst)[c] = ((const int4*)src)[c];
  }
  __syncthreads();

  const int o = blockIdx.y * 128 + wave * 16 + (lane & 15);
  const float* vrow = V + ((size_t)p * DOUT + o) * 256 + ((lane >> 4) << 4);
  const bf16* arow = zsh + (lane & 15) * 264 + ((lane >> 4) << 4);

  f32x8 acc[4] = {};
  float sq = 0.0f;

  for (int k0 = 0; k0 < 256; k0 += 32) {
    bf16x16 bfrag;
#pragma unroll
    for (int c = 0; c < 4; ++c) {
      float4 f = ((const float4*)(vrow + k0))[c];
      sq = fmaf(f.x, f.x, sq); sq = fmaf(f.y, f.y, sq);
      sq = fmaf(f.z, f.z, sq); sq = fmaf(f.w, f.w, sq);
      bfrag[4 * c + 0] = (bf16)f.x; bfrag[4 * c + 1] = (bf16)f.y;
      bfrag[4 * c + 2] = (bf16)f.z; bfrag[4 * c + 3] = (bf16)f.w;
    }
#pragma unroll
    for (int t = 0; t < 4; ++t) {
      const bf16* at = arow + (16 * t) * 264 + k0;
      bf16x8 lo = *(const bf16x8*)(at);
      bf16x8 hi = *(const bf16x8*)(at + 8);
      bf16x16 afrag;
#pragma unroll
      for (int i = 0; i < 8; ++i) { afrag[i] = lo[i]; afrag[8 + i] = hi[i]; }
      acc[t] = WMMA_BF16(afrag, bfrag, acc[t]);
    }
  }

  // combine the two K-halves of the sum of squares (lane L and L+16 share a column)
  const float tot = sq + __shfl_xor(sq, 16, 32);
  const float scale = G[(size_t)p * DOUT + o] * rsqrtf(fmaxf(tot, 1e-30f));

  const int bb = (lane >> 4) << 3;
#pragma unroll
  for (int t = 0; t < 4; ++t) {
#pragma unroll
    for (int i = 0; i < 8; ++i) {
      const int b = 16 * t + bb + i;
      Out[((size_t)b * PNUM + p) * DOUT + o] = acc[t][i] * scale;
    }
  }
}

// ---------------------------------------------------------------------------
extern "C" void kernel_launch(void* const* d_in, const int* in_sizes, int n_in,
                              void* d_out, int out_size, void* d_ws, size_t ws_size,
                              hipStream_t stream) {
  const float* x  = (const float*)d_in[0];
  const float* W1 = (const float*)d_in[1];
  const float* b1 = (const float*)d_in[2];
  const float* W2 = (const float*)d_in[3];
  const float* b2 = (const float*)d_in[4];
  const float* W3 = (const float*)d_in[5];
  const float* b3 = (const float*)d_in[6];
  const float* v  = (const float*)d_in[7];
  const float* g  = (const float*)d_in[8];
  float* out = (float*)d_out;

  char* ws = (char*)d_ws;
  size_t off = 0;
  auto alloc = [&](size_t bytes) -> char* {
    char* pp = ws + off;
    off = (off + bytes + 255) & ~(size_t)255;
    return pp;
  };
  bf16* xb  = (bf16*)alloc((size_t)MROWS * DIN * 2);
  bf16* w1b = (bf16*)alloc((size_t)DH * DIN * 2);
  bf16* w2b = (bf16*)alloc((size_t)DH * DH * 2);
  bf16* w3b = (bf16*)alloc((size_t)DBT * DH * 2);
  bf16* h1  = (bf16*)alloc((size_t)MROWS * DH * 2);
  bf16* h2  = (bf16*)alloc((size_t)MROWS * DH * 2);
  float* zf = (float*)alloc((size_t)MROWS * DBT * 4);
  bf16* zb  = (bf16*)alloc((size_t)MROWS * DBT * 2);
  (void)ws_size; (void)in_sizes; (void)n_in; (void)out_size;

  // 1) fp32 -> bf16 converts
  {
    int n4;
    n4 = MROWS * DIN / 4; cvt_f32_bf16<<<(n4 + 255) / 256, 256, 0, stream>>>(x,  xb,  n4);
    n4 = DH * DIN / 4;    cvt_f32_bf16<<<(n4 + 255) / 256, 256, 0, stream>>>(W1, w1b, n4);
    n4 = DH * DH / 4;     cvt_f32_bf16<<<(n4 + 255) / 256, 256, 0, stream>>>(W2, w2b, n4);
    n4 = DBT * DH / 4;    cvt_f32_bf16<<<(n4 + 255) / 256, 256, 0, stream>>>(W3, w3b, n4);
  }

  // 2) MLP GEMMs with WMMA (bf16 in, fp32 accumulate)
  dim3 g12(MROWS / 64, DH / 128);
  gemm_mlp<true, true><<<g12, 256, 0, stream>>>(xb, w1b, b1, (void*)h1, DIN, DH);
  gemm_mlp<true, true><<<g12, 256, 0, stream>>>(h1, w2b, b2, (void*)h2, DH, DH);
  dim3 g3(MROWS / 64, DBT / 128);
  gemm_mlp<false, false><<<g3, 256, 0, stream>>>(h2, w3b, b3, (void*)zf, DH, DBT);

  // 3) L2 normalize bottleneck rows -> bf16
  l2norm_rows<<<MROWS / 8, 256, 0, stream>>>(zf, zb);

  // 4) Fused weight-norm head (streams v exactly once)
  dim3 gh(PNUM, DOUT / 128);
  head_gemm<<<gh, 256, 0, stream>>>(zb, v, g, out);
}
